// QPNet_68126771249554
// MI455X (gfx1250) — compile-verified, matrix operand order checked by reading, not run on hardware
//
#include <hip/hip_runtime.h>
#include <hip/hip_bf16.h>

// ---------------------------------------------------------------------------
// QPNet (WaveNet-like) forward for MI455X / gfx1250, wave32 + WMMA f16.
//
//  - Activations ("actT", "gatedT"): time-major f16, pitch 544 halves
//    (544 = 68*8 -> 16B-aligned rows; 512 channels + pad), 2176 rows so padded
//    NT=4 tile reads stay in-bounds.
//  - hup: time-major f16, pitch 64 (39 aux channels zero-padded to 64), 2176 rows.
//  - Weights packed in WMMA A-fragment order: tile (mtile,ktile) = 512 halves;
//    consecutive ktiles are contiguous -> A pointers advance by +512/kt.
//  - B fragment (K x N, 16-bit): lane n = lane%16, K group g = 8*(lane/16);
//    halves 0..7 -> k0+g.., halves 8..15 -> k0+16+g.. (per CDNA5 ISA tables).
//  - NT = 4 time-tiles per wave: each A fragment feeds 4 WMMAs (4x less L2
//    weight traffic); per-j LDS/global offsets are compile-time immediates.
//  - 256-thread blocks (8 waves, 2/SIMD): better latency hiding, and 8 M-tiles
//    share one LDS slab staging.  __launch_bounds__(256,1) avoids spills.
//  - A-branch pitch gather folded into the LDS slab: idx[t] in [t, t+d] is
//    always inside the staged rows -> xP is a per-lane LDS row pointer.
// ---------------------------------------------------------------------------

typedef __attribute__((ext_vector_type(16))) _Float16 v16h;
typedef __attribute__((ext_vector_type(8)))  _Float16 v8h;
typedef __attribute__((ext_vector_type(8)))  float    v8f;

#define ACT_PITCH 544
#define ACT_ROWS  2176
#define HUP_PITCH 64
#define HUP_ROWS  2176
#define SLAB_PITCH 520
#define NT 4
#define SLAB_ROWS (NT * 16 + 8)
#define SLAB_BYTES (SLAB_ROWS * SLAB_PITCH * 2)
#define SKIP_SLAB_BYTES (NT * 16 * SLAB_PITCH * 2)

__device__ __forceinline__ v8f wmma_f16(v16h a, v16h b, v8f c) {
  return __builtin_amdgcn_wmma_f32_16x16x32_f16(false, a, false, b, (short)0, c,
                                                false, false);
}

__device__ __forceinline__ int fragK(int lane, int j) {
  int g = (lane >> 4) << 3;                 // 0 or 8
  return (j < 8) ? (g + j) : (16 + g + (j - 8));
}

union V16u { v16h v; v8h h[2]; };

__device__ __forceinline__ v16h ldFrag(const _Float16* p) {
  V16u b;
  b.h[0] = *(const v8h*)p;
  b.h[1] = *(const v8h*)(p + 16);
  return b.v;
}

// ---------------------------------------------------------------------------
// Weight packing: f32 source -> f16 WMMA-A-fragment order.
// src element (o, k) at  W[o*rowStride + k*kStride + off];  k >= Korig -> 0.
// ---------------------------------------------------------------------------
__global__ void pack_w_kernel(const float* __restrict__ W, _Float16* __restrict__ dst,
                              int M, int Kp, int Korig, int rowStride, int kStride,
                              int off) {
  int idx = blockIdx.x * blockDim.x + threadIdx.x;
  int total = M * Kp;
  if (idx >= total) return;
  int j = idx & 15;
  int lane = (idx >> 4) & 31;
  int tile = idx >> 9;
  int ktiles = Kp >> 5;
  int ktile = tile % ktiles;
  int mtile = tile / ktiles;
  int o = mtile * 16 + (lane & 15);
  int k = ktile * 32 + fragK(lane, j);
  float v = 0.f;
  if (k < Korig && o < M)
    v = W[(size_t)o * rowStride + (size_t)k * kStride + off];
  dst[idx] = (_Float16)v;
}

// actT[t][o] = Wc[o][x[t]][0] + Wc[o][x[t+1]][1] + bc[o]
__global__ void embed_kernel(const int* __restrict__ x, const float* __restrict__ Wc,
                             const float* __restrict__ bc, _Float16* __restrict__ actT) {
  int t = blockIdx.x;                       // 0..2059
  int o = threadIdx.x;                      // 0..511
  int i0 = x[t] & 255;
  int i1 = x[t + 1] & 255;
  float v = Wc[(size_t)o * 512 + i0 * 2] + Wc[(size_t)o * 512 + i1 * 2 + 1] + bc[o];
  actT[(size_t)t * ACT_PITCH + o] = (_Float16)v;
}

// hupT[j][a] = h[a][j/110]*Wup[j%110] + bup ; zero-pad a>=39 or j>=2090
__global__ void hup_kernel(const float* __restrict__ h, const float* __restrict__ wup,
                           const float* __restrict__ bup, _Float16* __restrict__ hupT) {
  int j = blockIdx.x;
  int a = threadIdx.x;
  float v = 0.f;
  if (a < 39 && j < 2090)
    v = h[a * 19 + (j / 110)] * wup[j % 110] + bup[0];
  hupT[(size_t)j * HUP_PITCH + a] = (_Float16)v;
}

__global__ void zero_f32_kernel(float* p, int n) {
  int i = blockIdx.x * blockDim.x + threadIdx.x;
  if (i < n) p[i] = 0.f;
}

__global__ void relu_f32_to_f16_kernel(const float* __restrict__ src,
                                       _Float16* __restrict__ dst, int n) {
  int i = blockIdx.x * blockDim.x + threadIdx.x;
  if (i < n) { float v = src[i]; dst[i] = (_Float16)(v > 0.f ? v : 0.f); }
}

// idx[n] = clamp(shift + n - round(d*df[dfStart+n]), 0, Lprev-1); idx in [n, n+d]
__global__ void aidx_kernel(const float* __restrict__ df, int dfStart, int shift,
                            int d, int Lg, int Lprev, int* __restrict__ idx) {
  int n = blockIdx.x * blockDim.x + threadIdx.x;
  if (n >= Lg) return;
  int v = shift + n - (int)rintf((float)d * df[dfStart + n]);
  idx[n] = v < 0 ? 0 : (v >= Lprev ? Lprev - 1 : v);
}

// ---------------------------------------------------------------------------
// Gate kernel (templated on FMODE):
//  S = W0S@B0 + W1S@B1 + WxS@hup + biases ; same for T; gated = sigmoid(S)*tanh(T)
// FMODE=1: B0 = actT[t],        B1 = actT[t + dshift]
// FMODE=0: B0 = actT[t+dshift], B1 = actT[idx[t]]   (xP row is inside the slab)
// Block: 256 threads (8 waves); grid (ceil(Lout/64), 4) -> 32 M-tiles of 16.
// ---------------------------------------------------------------------------
template <int FMODE>
__launch_bounds__(256, 1)
__global__ void gate_kernel(
    const _Float16* __restrict__ actT, int Lin,
    const int* __restrict__ idxBuf, int dshift,
    const _Float16* __restrict__ hupT, int hOff,
    const _Float16* __restrict__ pW0S, const _Float16* __restrict__ pW1S,
    const _Float16* __restrict__ pWxS,
    const _Float16* __restrict__ pW0T, const _Float16* __restrict__ pW1T,
    const _Float16* __restrict__ pWxT,
    const float* __restrict__ bS1, const float* __restrict__ bS2,
    const float* __restrict__ bS3,
    const float* __restrict__ bT1, const float* __restrict__ bT2,
    const float* __restrict__ bT3,
    _Float16* __restrict__ gatedT, int Lout) {
  extern __shared__ _Float16 slab[];

  int tid = threadIdx.x;
  int wave = tid >> 5, lane = tid & 31;
  int t0 = blockIdx.x * (16 * NT);

  // Cooperative staging: rows t0 .. t0 + NT*16 + dshift - 1 (zero-padded).
  int nR = NT * 16 + dshift;
  for (int i = tid; i < nR * 64; i += 256) {
    int r = i >> 6, c8 = (i & 63) << 3;
    int t = t0 + r;
    v8h v = {};
    if (t < Lin)
      v = *(const v8h*)(actT + (size_t)t * ACT_PITCH + c8);
    *(v8h*)(slab + r * SLAB_PITCH + c8) = v;
  }
  __syncthreads();

  int mtile = blockIdx.y * 8 + wave;
  int n = lane & 15, hi = lane >> 4, g = hi << 3;

  // A-fragment base pointers; consecutive ktiles contiguous -> += 512 per kt.
  size_t wb0 = (((size_t)mtile * 16) << 9) + ((size_t)lane << 4);
  const _Float16* pa0s = pW0S + wb0;
  const _Float16* pa1s = pW1S + wb0;
  const _Float16* pa0t = pW0T + wb0;
  const _Float16* pa1t = pW1T + wb0;

  // B base pointers (k0 folded in by += 32 per kt; per-j offsets are immediates).
  const _Float16* pb0 = slab + (FMODE ? n : n + dshift) * SLAB_PITCH + g;
  const _Float16* pb1f = pb0 + dshift * SLAB_PITCH;      // FMODE==1 path
  const _Float16* pb1a[NT];                              // FMODE==0 path
  if (!FMODE) {
#pragma unroll
    for (int j = 0; j < NT; ++j) {
      int t = t0 + 16 * j + n;
      int row = 0;
      if (t < Lout) row = idxBuf[t] - t0;  // guaranteed in [16j+n, 16j+n+dshift]
      pb1a[j] = slab + row * SLAB_PITCH + g;
    }
  }

  v8f accS[NT], accT[NT];
#pragma unroll
  for (int j = 0; j < NT; ++j) { accS[j] = (v8f){}; accT[j] = (v8f){}; }

  for (int kt = 0; kt < 16; ++kt) {
    v16h a0s = *(const v16h*)pa0s;
    v16h a1s = *(const v16h*)pa1s;
    v16h a0t = *(const v16h*)pa0t;
    v16h a1t = *(const v16h*)pa1t;
#pragma unroll
    for (int j = 0; j < NT; ++j) {
      v16h b0 = ldFrag(pb0 + j * 16 * SLAB_PITCH);
      v16h b1 = FMODE ? ldFrag(pb1f + j * 16 * SLAB_PITCH) : ldFrag(pb1a[j]);
      accS[j] = wmma_f16(a0s, b0, accS[j]);
      accS[j] = wmma_f16(a1s, b1, accS[j]);
      accT[j] = wmma_f16(a0t, b0, accT[j]);
      accT[j] = wmma_f16(a1t, b1, accT[j]);
    }
    pa0s += 512; pa1s += 512; pa0t += 512; pa1t += 512;
    pb0 += 32; pb1f += 32;
    if (!FMODE) {
#pragma unroll
      for (int j = 0; j < NT; ++j) pb1a[j] += 32;
    }
  }

  // Aux features (K padded 39 -> 64), B directly from zero-padded global hup.
  size_t wbx = (((size_t)mtile * 2) << 9) + ((size_t)lane << 4);
  const _Float16* ph0 = hupT + (size_t)(hOff + t0 + n) * HUP_PITCH + g;
#pragma unroll
  for (int kt = 0; kt < 2; ++kt) {
    v16h axs = *(const v16h*)(pWxS + wbx + (size_t)kt * 512);
    v16h axt = *(const v16h*)(pWxT + wbx + (size_t)kt * 512);
#pragma unroll
    for (int j = 0; j < NT; ++j) {
      v16h bh = ldFrag(ph0 + kt * 32 + (size_t)j * 16 * HUP_PITCH);
      accS[j] = wmma_f16(axs, bh, accS[j]);
      accT[j] = wmma_f16(axt, bh, accT[j]);
    }
  }

  int ob = mtile * 16 + g;                  // C/D rows ob..ob+7 (VGPR r -> M=r+8*hi)
  float cbS[8], cbT[8];
#pragma unroll
  for (int r = 0; r < 8; ++r) {
    int o = ob + r;
    cbS[r] = bS1[o] + bS2[o] + (FMODE ? 0.f : bS3[o]);
    cbT[r] = bT1[o] + bT2[o] + (FMODE ? 0.f : bT3[o]);
  }
#pragma unroll
  for (int j = 0; j < NT; ++j) {
    int t = t0 + 16 * j + n;
    if (t >= Lout) continue;
    v8h outv;
#pragma unroll
    for (int r = 0; r < 8; ++r) {
      float s = accS[j][r] + cbS[r];
      float tt = accT[j][r] + cbT[r];
      outv[r] = (_Float16)(tanhf(tt) / (1.f + __expf(-s)));
    }
    *(v8h*)(gatedT + (size_t)t * ACT_PITCH + ob) = outv;
  }
}

// ---------------------------------------------------------------------------
// Skip + residual projections from gated (NT=4, LDS-staged B):
//  mtile < 16 : skipAcc[t'][o] += (Wskip @ gated)[o][t] + bskip   (last 2000 t)
//  mtile >=16 : actNextT[t][o]  = (Wres  @ gated)[o][t] + bres + actPrevT[t+d][o]
// Block: 256 threads (8 waves); grid (ceil(Lout/64), 6) -> 48 M-tiles of 16.
// ---------------------------------------------------------------------------
__launch_bounds__(256, 1)
__global__ void skipres_kernel(
    const _Float16* __restrict__ gatedT, int Lout,
    const _Float16* __restrict__ pWskip, const float* __restrict__ bskip,
    float* __restrict__ skipAcc,
    const _Float16* __restrict__ pWres, const float* __restrict__ bres,
    const _Float16* __restrict__ actPrevT, int d,
    _Float16* __restrict__ actNextT) {
  extern __shared__ _Float16 slab[];

  int tid = threadIdx.x;
  int wave = tid >> 5, lane = tid & 31;
  int t0 = blockIdx.x * (16 * NT);

  // Stage the gated tile (64 rows) once per block.
  for (int i = tid; i < NT * 16 * 64; i += 256) {
    int r = i >> 6, c8 = (i & 63) << 3;
    int t = t0 + r;
    v8h v = {};
    if (t < Lout)
      v = *(const v8h*)(gatedT + (size_t)t * ACT_PITCH + c8);
    *(v8h*)(slab + r * SLAB_PITCH + c8) = v;
  }
  __syncthreads();

  int mtile = blockIdx.y * 8 + wave;
  int n = lane & 15, hi = lane >> 4, g = hi << 3;
  bool isSkip = mtile < 16;
  int mt = isSkip ? mtile : (mtile - 16);

  const _Float16* pa = (isSkip ? pWskip : pWres) +
                       (((size_t)mt * 16) << 9) + ((size_t)lane << 4);
  const _Float16* pb = slab + n * SLAB_PITCH + g;

  v8f acc[NT];
#pragma unroll
  for (int j = 0; j < NT; ++j) acc[j] = (v8f){};

  for (int kt = 0; kt < 16; ++kt) {
    v16h a = *(const v16h*)pa;
#pragma unroll
    for (int j = 0; j < NT; ++j) {
      v16h b = ldFrag(pb + j * 16 * SLAB_PITCH);
      acc[j] = wmma_f16(a, b, acc[j]);
    }
    pa += 512;
    pb += 32;
  }

  int ob = mt * 16 + g;
#pragma unroll
  for (int j = 0; j < NT; ++j) {
    int t = t0 + 16 * j + n;
    if (t >= Lout) continue;
    if (isSkip) {
      int ts = t - (Lout - 2000);
      if (ts >= 0) {
#pragma unroll
        for (int r = 0; r < 8; ++r) {
          int o = ob + r;
          skipAcc[(size_t)ts * 256 + o] += acc[j][r] + bskip[o];
        }
      }
    } else {
      v8h pv = *(const v8h*)(actPrevT + (size_t)(t + d) * ACT_PITCH + ob);
      v8h outv;
#pragma unroll
      for (int r = 0; r < 8; ++r) {
        int o = ob + r;
        outv[r] = (_Float16)(acc[j][r] + bres[o] + (float)pv[r]);
      }
      *(v8h*)(actNextT + (size_t)t * ACT_PITCH + ob) = outv;
    }
  }
}

// ---------------------------------------------------------------------------
// Post GEMM: C = W(256x256) @ B(256xN) + bias ; optional relu. NT=4 tiles.
// mode 0: write f16 [t*256+o] ; mode 1: write f32 [t*256+o] (final output).
// Block: 256 threads (8 waves); grid (ceil(N/64), 2) -> 16 M-tiles of 16.
// ---------------------------------------------------------------------------
__launch_bounds__(256, 1)
__global__ void gemm_post_kernel(const _Float16* __restrict__ pW,
                                 const _Float16* __restrict__ srcT,
                                 const float* __restrict__ bias, int N,
                                 int reluOut, int mode,
                                 _Float16* __restrict__ dstT,
                                 float* __restrict__ dstF) {
  int tid = threadIdx.x;
  int wave = tid >> 5, lane = tid & 31;
  int t0 = blockIdx.x * (16 * NT);
  int mtile = blockIdx.y * 8 + wave;
  int n = lane & 15, hi = lane >> 4, g = hi << 3;

  const _Float16* pa = pW + (((size_t)mtile * 8) << 9) + ((size_t)lane << 4);
  const _Float16* pb = srcT + (size_t)(t0 + n) * 256 + g;

  v8f acc[NT];
#pragma unroll
  for (int j = 0; j < NT; ++j) acc[j] = (v8f){};

#pragma unroll
  for (int kt = 0; kt < 8; ++kt) {
    v16h a = *(const v16h*)(pa + (size_t)kt * 512);
#pragma unroll
    for (int j = 0; j < NT; ++j) {
      v16h b = ldFrag(pb + kt * 32 + (size_t)j * 16 * 256);
      acc[j] = wmma_f16(a, b, acc[j]);
    }
  }
  int ob = mtile * 16 + g;
#pragma unroll
  for (int j = 0; j < NT; ++j) {
    int t = t0 + 16 * j + n;
    if (t >= N) continue;
    if (mode == 0) {
      v8h outv;
#pragma unroll
      for (int r = 0; r < 8; ++r) {
        float v = acc[j][r] + bias[ob + r];
        if (reluOut) v = v > 0.f ? v : 0.f;
        outv[r] = (_Float16)v;
      }
      *(v8h*)(dstT + (size_t)t * 256 + ob) = outv;
    } else {
#pragma unroll
      for (int r = 0; r < 8; ++r) {
        float v = acc[j][r] + bias[ob + r];
        if (reluOut) v = v > 0.f ? v : 0.f;
        dstF[(size_t)t * 256 + ob + r] = v;
      }
    }
  }
}

// ---------------------------------------------------------------------------
// Host orchestration
// ---------------------------------------------------------------------------
extern "C" void kernel_launch(void* const* d_in, const int* in_sizes, int n_in,
                              void* d_out, int out_size, void* d_ws, size_t ws_size,
                              hipStream_t stream) {
  (void)in_sizes; (void)n_in; (void)out_size; (void)ws_size;

  const int*   x        = (const int*)d_in[0];
  const float* h        = (const float*)d_in[1];
  const float* df       = (const float*)d_in[2];
  const float* Wc       = (const float*)d_in[4];
  const float* bc       = (const float*)d_in[5];
  const float* Wup      = (const float*)d_in[6];
  const float* bup      = (const float*)d_in[7];
  const float* F_WdilS  = (const float*)d_in[8];
  const float* F_bdilS  = (const float*)d_in[9];
  const float* F_WdilT  = (const float*)d_in[10];
  const float* F_bdilT  = (const float*)d_in[11];
  const float* F_WauxS  = (const float*)d_in[12];
  const float* F_bauxS  = (const float*)d_in[13];
  const float* F_WauxT  = (const float*)d_in[14];
  const float* F_bauxT  = (const float*)d_in[15];
  const float* F_Wskip  = (const float*)d_in[16];
  const float* F_bskip  = (const float*)d_in[17];
  const float* F_Wres   = (const float*)d_in[18];
  const float* F_bres   = (const float*)d_in[19];
  const float* A_WsigC  = (const float*)d_in[20];
  const float* A_bsigC  = (const float*)d_in[21];
  const float* A_WsigP  = (const float*)d_in[22];
  const float* A_bsigP  = (const float*)d_in[23];
  const float* A_WtanC  = (const float*)d_in[24];
  const float* A_btanC  = (const float*)d_in[25];
  const float* A_WtanP  = (const float*)d_in[26];
  const float* A_btanP  = (const float*)d_in[27];
  const float* A_WauxS  = (const float*)d_in[28];
  const float* A_bauxS  = (const float*)d_in[29];
  const float* A_WauxT  = (const float*)d_in[30];
  const float* A_bauxT  = (const float*)d_in[31];
  const float* A_Wskip  = (const float*)d_in[32];
  const float* A_bskip  = (const float*)d_in[33];
  const float* A_Wres   = (const float*)d_in[34];
  const float* A_bres   = (const float*)d_in[35];
  const float* W_post1  = (const float*)d_in[36];
  const float* b_post1  = (const float*)d_in[37];
  const float* W_post2  = (const float*)d_in[38];
  const float* b_post2  = (const float*)d_in[39];

  char* ws = (char*)d_ws;
  size_t off = 0;
  auto bump = [&](size_t bytes) -> char* {
    char* p = ws + off;
    off += (bytes + 255) & ~(size_t)255;
    return p;
  };

  const size_t actBytes = (size_t)ACT_ROWS * ACT_PITCH * sizeof(_Float16);
  _Float16* actA   = (_Float16*)bump(actBytes);
  _Float16* actB   = (_Float16*)bump(actBytes);
  _Float16* gatedT = (_Float16*)bump(actBytes);
  _Float16* hupT   = (_Float16*)bump((size_t)HUP_ROWS * HUP_PITCH * sizeof(_Float16));
  float*    skipAcc= (float*)bump((size_t)2000 * 256 * sizeof(float));
  _Float16* p0     = (_Float16*)bump((size_t)2048 * 256 * sizeof(_Float16));
  _Float16* p1     = (_Float16*)bump((size_t)2048 * 256 * sizeof(_Float16));
  int*      idxBuf = (int*)bump((size_t)2080 * sizeof(int));

  auto pb16 = [&](size_t halves) -> _Float16* {
    return (_Float16*)bump(halves * sizeof(_Float16));
  };

  _Float16 *pF_S0[12], *pF_S1[12], *pF_T0[12], *pF_T1[12];
  _Float16 *pF_xS[12], *pF_xT[12], *pF_sk[12], *pF_rs[12];
  for (int l = 0; l < 12; ++l) {
    pF_S0[l] = pb16(512 * 512); pF_S1[l] = pb16(512 * 512);
    pF_T0[l] = pb16(512 * 512); pF_T1[l] = pb16(512 * 512);
    pF_xS[l] = pb16(512 * 64);  pF_xT[l] = pb16(512 * 64);
    pF_sk[l] = pb16(256 * 512); pF_rs[l] = pb16(512 * 512);
  }
  _Float16 *pA_sC[4], *pA_sP[4], *pA_tC[4], *pA_tP[4];
  _Float16 *pA_xS[4], *pA_xT[4], *pA_sk[4], *pA_rs[4];
  for (int l = 0; l < 4; ++l) {
    pA_sC[l] = pb16(512 * 512); pA_sP[l] = pb16(512 * 512);
    pA_tC[l] = pb16(512 * 512); pA_tP[l] = pb16(512 * 512);
    pA_xS[l] = pb16(512 * 64);  pA_xT[l] = pb16(512 * 64);
    pA_sk[l] = pb16(256 * 512); pA_rs[l] = pb16(512 * 512);
  }
  _Float16* pP1 = pb16(256 * 256);
  _Float16* pP2 = pb16(256 * 256);

  auto pack = [&](const float* W, _Float16* dst, int M, int Kp, int Korig,
                  int rowStride, int kStride, int o) {
    int total = M * Kp;
    pack_w_kernel<<<(total + 255) / 256, 256, 0, stream>>>(
        W, dst, M, Kp, Korig, rowStride, kStride, o);
  };

  for (int l = 0; l < 12; ++l) {
    const float* ws_ = F_WdilS + (size_t)l * 512 * 512 * 2;
    const float* wt_ = F_WdilT + (size_t)l * 512 * 512 * 2;
    pack(ws_, pF_S0[l], 512, 512, 512, 1024, 2, 0);
    pack(ws_, pF_S1[l], 512, 512, 512, 1024, 2, 1);
    pack(wt_, pF_T0[l], 512, 512, 512, 1024, 2, 0);
    pack(wt_, pF_T1[l], 512, 512, 512, 1024, 2, 1);
    pack(F_WauxS + (size_t)l * 512 * 39, pF_xS[l], 512, 64, 39, 39, 1, 0);
    pack(F_WauxT + (size_t)l * 512 * 39, pF_xT[l], 512, 64, 39, 39, 1, 0);
    pack(F_Wskip + (size_t)l * 256 * 512, pF_sk[l], 256, 512, 512, 512, 1, 0);
    pack(F_Wres  + (size_t)l * 512 * 512, pF_rs[l], 512, 512, 512, 512, 1, 0);
  }
  for (int l = 0; l < 4; ++l) {
    pack(A_WsigC + (size_t)l * 512 * 512, pA_sC[l], 512, 512, 512, 512, 1, 0);
    pack(A_WsigP + (size_t)l * 512 * 512, pA_sP[l], 512, 512, 512, 512, 1, 0);
    pack(A_WtanC + (size_t)l * 512 * 512, pA_tC[l], 512, 512, 512, 512, 1, 0);
    pack(A_WtanP + (size_t)l * 512 * 512, pA_tP[l], 512, 512, 512, 512, 1, 0);
    pack(A_WauxS + (size_t)l * 512 * 39, pA_xS[l], 512, 64, 39, 39, 1, 0);
    pack(A_WauxT + (size_t)l * 512 * 39, pA_xT[l], 512, 64, 39, 39, 1, 0);
    pack(A_Wskip + (size_t)l * 256 * 512, pA_sk[l], 256, 512, 512, 512, 1, 0);
    pack(A_Wres  + (size_t)l * 512 * 512, pA_rs[l], 512, 512, 512, 512, 1, 0);
  }
  pack(W_post1, pP1, 256, 256, 256, 256, 1, 0);
  pack(W_post2, pP2, 256, 256, 256, 256, 1, 0);

  zero_f32_kernel<<<(512000 + 255) / 256, 256, 0, stream>>>(skipAcc, 512000);
  embed_kernel<<<2060, 512, 0, stream>>>(x, Wc, bc, actA);
  hup_kernel<<<HUP_ROWS, 64, 0, stream>>>(h, Wup, bup, hupT);

  _Float16* cur = actA;
  _Float16* nxt = actB;
  int Lcur = 2060;

  static const int DF[12] = {1, 2, 4, 8, 1, 2, 4, 8, 1, 2, 4, 8};
  for (int l = 0; l < 12; ++l) {
    int d = DF[l];
    int Lout = Lcur - d;
    int nT4 = (Lout + 16 * NT - 1) / (16 * NT);
    gate_kernel<1><<<dim3(nT4, 4), 256, SLAB_BYTES, stream>>>(
        cur, /*Lin=*/Lcur, /*idxBuf=*/nullptr, /*dshift=*/d,
        hupT, /*hOff=*/2090 - Lout,
        pF_S0[l], pF_S1[l], pF_xS[l], pF_T0[l], pF_T1[l], pF_xT[l],
        F_bdilS + l * 512, F_bauxS + l * 512, nullptr,
        F_bdilT + l * 512, F_bauxT + l * 512, nullptr,
        gatedT, Lout);
    skipres_kernel<<<dim3(nT4, 6), 256, SKIP_SLAB_BYTES, stream>>>(
        gatedT, Lout, pF_sk[l], F_bskip + l * 256, skipAcc,
        pF_rs[l], F_bres + l * 512, cur, d, nxt);
    _Float16* t = cur; cur = nxt; nxt = t;
    Lcur = Lout;
  }

  static const int DA[4] = {1, 2, 4, 8};
  for (int l = 0; l < 4; ++l) {
    int d = DA[l];
    int Lg = Lcur - d;
    int nT4 = (Lg + 16 * NT - 1) / (16 * NT);
    aidx_kernel<<<(Lg + 255) / 256, 256, 0, stream>>>(
        df, /*dfStart=*/2061 - Lg, /*shift=*/d, d, Lg, Lcur, idxBuf);
    gate_kernel<0><<<dim3(nT4, 4), 256, SLAB_BYTES, stream>>>(
        cur, /*Lin=*/Lcur, idxBuf, /*dshift=*/d,
        hupT, /*hOff=*/2090 - Lg,
        pA_sC[l], pA_sP[l], pA_xS[l], pA_tC[l], pA_tP[l], pA_xT[l],
        A_bsigC + l * 512, A_bsigP + l * 512, A_bauxS + l * 512,
        A_btanC + l * 512, A_btanP + l * 512, A_bauxT + l * 512,
        gatedT, Lg);
    skipres_kernel<<<dim3(nT4, 6), 256, SKIP_SLAB_BYTES, stream>>>(
        gatedT, Lg, pA_sk[l], A_bskip + l * 256, skipAcc,
        pA_rs[l], A_bres + l * 512, cur, d, nxt);
    _Float16* t = cur; cur = nxt; nxt = t;
    Lcur = Lg;
  }

  relu_f32_to_f16_kernel<<<(512000 + 255) / 256, 256, 0, stream>>>(skipAcc, p0, 512000);
  gemm_post_kernel<<<dim3((2000 + 63) / 64, 2), 256, 0, stream>>>(
      pP1, p0, b_post1, 2000, /*reluOut=*/1, /*mode=*/0, p1, nullptr);
  gemm_post_kernel<<<dim3((2000 + 63) / 64, 2), 256, 0, stream>>>(
      pP2, p1, b_post2, 2000, /*reluOut=*/0, /*mode=*/1, nullptr, (float*)d_out);
}
